// Decoder_39625368273304
// MI455X (gfx1250) — matrix operand order repaired and checked by reading, compiled, not run on hardware
//
#include <hip/hip_runtime.h>
#include <hip/hip_bf16.h>

// ---------------------------------------------------------------------------
// Problem constants (from reference)
// ---------------------------------------------------------------------------
#define NB   32            // batch images
#define NP   300000        // points
#define NL   64            // Zernike basis size
#define NIMG 256           // image box
#define IMG_PIX (NIMG*NIMG)
#define TILES (NP/16)      // 18750 point-tiles of 16

typedef __attribute__((ext_vector_type(16))) _Float16 v16h;
typedef __attribute__((ext_vector_type(8)))  _Float16 v8h;
typedef __attribute__((ext_vector_type(8)))  float    v8f;
typedef __attribute__((ext_vector_type(4)))  unsigned int u32x4;
typedef __attribute__((ext_vector_type(4)))  int      i32x4;
typedef __attribute__((ext_vector_type(8)))  int      i32x8;

static __device__ __forceinline__ v8f wmma16x16x32_f16(v16h a, v16h b, v8f c) {
    // D = A(16x32 f16) * B(32x16 f16) + C(16x16 f32)
    return __builtin_amdgcn_wmma_f32_16x16x32_f16(
        /*neg_a=*/false, a, /*neg_b=*/false, b,
        /*c_mod=*/(short)0, c, /*reuse_a=*/false, /*reuse_b=*/false);
}

static __device__ __forceinline__ int iclamp(int v, int lo, int hi) {
    return v < lo ? lo : (v > hi ? hi : v);
}

// ---------------------------------------------------------------------------
// TDM: DMA one contiguous run of 2048 f32 (8 KB) from global memory into LDS.
// D# per CDNA5 ISA:
//   group0: [1:0]=count=1, [63:32]=lds_addr, [120:64]=global_addr, [127:126]=2
//   group1: data_size=4B, tensor_dim0=2048, tensor_dim1=1, tile_dim0=2048,
//           tile_dim1=1, tensor_dim0_stride=2048
//   groups 2/3: zero (tensor <= 2D)
// Tracked by TENSORcnt; EXEC is ignored (wave-level op).
// ---------------------------------------------------------------------------
static __device__ __forceinline__ void tdm_load_8kb(void* ldsDst, const void* gSrc)
{
    const unsigned lds_addr = (unsigned)(uintptr_t)ldsDst;   // addr[31:0] = LDS byte addr
    const unsigned long long ga = (unsigned long long)(uintptr_t)gSrc;

    u32x4 g0;
    g0.x = 1u;                                               // count=1, user descriptor
    g0.y = lds_addr;
    g0.z = (unsigned)(ga & 0xFFFFFFFFull);
    g0.w = (unsigned)((ga >> 32) & 0x01FFFFFFull) | (2u << 30); // type=2 ("image")

    i32x8 g1;
    g1[0] = (int)(2u << 16);        // workgroup_mask=0, data_size=2 (4 bytes)
    g1[1] = (int)(2048u << 16);     // tensor_dim0[15:0] = 2048 in bits [63:48]
    g1[2] = (int)(1u << 16);        // tensor_dim0 hi = 0; tensor_dim1 = 1
    g1[3] = (int)(2048u << 16);     // tensor_dim1 hi = 0; tile_dim0 = 2048
    g1[4] = 1;                      // tile_dim1 = 1, tile_dim2 = 0
    g1[5] = 2048;                   // tensor_dim0_stride = 2048 (data_size units)
    g1[6] = 0;                      // stride hi, tensor_dim1_stride lo
    g1[7] = 0;                      // tensor_dim1_stride hi

    i32x4 z4 = {0, 0, 0, 0};
#if defined(__clang_major__) && (__clang_major__ >= 23)
    i32x8 z8 = {0, 0, 0, 0, 0, 0, 0, 0};
    __builtin_amdgcn_tensor_load_to_lds(g0, g1, z4, z4, z8, 0);
#else
    __builtin_amdgcn_tensor_load_to_lds(g0, g1, z4, z4, 0);
#endif
}

// ---------------------------------------------------------------------------
// Kernel 0: zero the scatter accumulator
// ---------------------------------------------------------------------------
__global__ void zero_buf(float* __restrict__ p, int n) {
    int i = blockIdx.x * blockDim.x + threadIdx.x;
    if (i < n) p[i] = 0.0f;
}

// ---------------------------------------------------------------------------
// Kernel 1: fused  d = z @ Z^T (WMMA f16)  ->  deform -> rotate -> shift
//                  -> bilinear scatter-splat with global f32 atomics.
// z matrices staged to LDS via the Tensor Data Mover; one wave handles
// 16 points x all 32 images (two 16-row M tiles).
// ---------------------------------------------------------------------------
__global__ __launch_bounds__(256) void splat_wmma(
    const float* __restrict__ zx, const float* __restrict__ zy,
    const float* __restrict__ zz, const float* __restrict__ Zb,
    const float* __restrict__ coords, const float* __restrict__ wts,
    const float* __restrict__ Rm, const float* __restrict__ shifts,
    float* __restrict__ img)
{
    __shared__ float    ldsZf32[3 * NB * NL];  // TDM destination (24 KB, f32)
    __shared__ _Float16 ldsZ[3][NB * NL];      // f16 WMMA staging (12 KB)
    __shared__ float    ldsR[NB * 9];          // rotation matrices
    __shared__ float    ldsS[NB * 2];          // shifts

    const int tid  = threadIdx.x;
    const int wave = tid >> 5;
    const int lane = tid & 31;

    // --- stage R/shifts with VMEM while the TDM streams z into LDS --------
    for (int i = tid; i < NB * 9; i += 256) ldsR[i] = Rm[i];
    if (tid < NB * 2) ldsS[tid] = shifts[tid];

    if (wave == 0) {   // TDM ops are wave-level (EXEC ignored); one wave issues
        tdm_load_8kb(&ldsZf32[0 * NB * NL], zx);
        tdm_load_8kb(&ldsZf32[1 * NB * NL], zy);
        tdm_load_8kb(&ldsZf32[2 * NB * NL], zz);
        __builtin_amdgcn_s_wait_tensorcnt((short)0);
    }
    __syncthreads();

    // --- convert staged z to f16 in the ISA A-fragment source layout ------
    for (int i = tid; i < 3 * NB * NL; i += 256) {
        int ax = i >> 11;          // / 2048
        int rem = i & 2047;
        ldsZ[ax][rem] = (_Float16)ldsZf32[i];
    }
    __syncthreads();

    const int tile = blockIdx.x * 8 + wave;
    if (tile >= TILES) return;               // uniform per wave: EXEC stays full

    const int pt0 = tile * 16;
    const int col = lane & 15;               // point column / A row (M)
    const int kh  = lane >> 4;               // lane-half selector per ISA layout
    const int pt  = pt0 + col;

    // prefetch next tiles' Z rows (speculative, L2)
    __builtin_prefetch(Zb + (size_t)(pt + 128) * NL, 0, 1);

    v8f accX[2], accY[2], accZ[2];
    {
        v8f z8 = {0.f,0.f,0.f,0.f,0.f,0.f,0.f,0.f};
        accX[0] = z8; accX[1] = z8;
        accY[0] = z8; accY[1] = z8;
        accZ[0] = z8; accZ[1] = z8;
    }

    #pragma unroll
    for (int kc = 0; kc < 2; ++kc) {         // K chunks of 32 (L = 64)
        // ---- B tile: B[k][n] = Z[pt=n][k];  element i -> K = kc*32 + kh*16 + i
        const float* Zp = Zb + (size_t)pt * NL + kc * 32 + kh * 16;
        v16h bv;
        #pragma unroll
        for (int i = 0; i < 16; ++i) bv[i] = (_Float16)Zp[i];

        #pragma unroll
        for (int mt = 0; mt < 2; ++mt) {     // image M tiles (rows 0-15, 16-31)
            const int row  = mt * 16 + col;                 // A row = image id
            const int base = row * NL + kc * 32 + kh * 16;  // halfword index
            // A layout: element i -> K = kc*32 + kh*16 + i
            {
                v8h lo = *(const v8h*)&ldsZ[0][base];
                v8h hi = *(const v8h*)&ldsZ[0][base + 8];
                v16h a = __builtin_shufflevector(lo, hi,
                        0,1,2,3,4,5,6,7,8,9,10,11,12,13,14,15);
                accX[mt] = wmma16x16x32_f16(a, bv, accX[mt]);
            }
            {
                v8h lo = *(const v8h*)&ldsZ[1][base];
                v8h hi = *(const v8h*)&ldsZ[1][base + 8];
                v16h a = __builtin_shufflevector(lo, hi,
                        0,1,2,3,4,5,6,7,8,9,10,11,12,13,14,15);
                accY[mt] = wmma16x16x32_f16(a, bv, accY[mt]);
            }
            {
                v8h lo = *(const v8h*)&ldsZ[2][base];
                v8h hi = *(const v8h*)&ldsZ[2][base + 8];
                v16h a = __builtin_shufflevector(lo, hi,
                        0,1,2,3,4,5,6,7,8,9,10,11,12,13,14,15);
                accZ[mt] = wmma16x16x32_f16(a, bv, accZ[mt]);
            }
        }
    }

    // ---- epilogue: one point per lane across all 32 images ---------------
    const float c0 = coords[(size_t)pt * 3 + 0];
    const float c1 = coords[(size_t)pt * 3 + 1];
    const float c2 = coords[(size_t)pt * 3 + 2];
    const float w  = wts[pt];
    const int rbase = kh * 8;   // C/D layout: lanes 0-15 -> M=r, lanes 16-31 -> M=8+r

    #pragma unroll
    for (int mt = 0; mt < 2; ++mt) {
        #pragma unroll
        for (int r = 0; r < 8; ++r) {
            const int im = mt * 16 + rbase + r;
            const float cx = c0 + accX[mt][r];
            const float cy = c1 + accY[mt][r];
            const float cz = c2 + accZ[mt][r];
            const float* Rr = &ldsR[im * 9];
            const float px = Rr[0]*cx + Rr[1]*cy + Rr[2]*cz + ldsS[im*2+0] + 128.0f;
            const float py = Rr[3]*cx + Rr[4]*cy + Rr[5]*cz + ldsS[im*2+1] + 128.0f;

            const float x0f = floorf(px), y0f = floorf(py);
            const float fx = px - x0f,    fy = py - y0f;
            const int xi0 = iclamp((int)x0f, 0, NIMG - 1);
            const int yi0 = iclamp((int)y0f, 0, NIMG - 1);
            const int xi1 = iclamp(xi0 + 1, 0, NIMG - 1);
            const int yi1 = iclamp(yi0 + 1, 0, NIMG - 1);

            float* ib = img + (size_t)im * IMG_PIX;
            atomicAdd(ib + yi0 * NIMG + xi0, w * (1.f - fx) * (1.f - fy));
            atomicAdd(ib + yi0 * NIMG + xi1, w * fx * (1.f - fy));
            atomicAdd(ib + yi1 * NIMG + xi0, w * (1.f - fx) * fy);
            atomicAdd(ib + yi1 * NIMG + xi1, w * fx * fy);
        }
    }
}

// ---------------------------------------------------------------------------
// Kernel 2: 5x5 Gaussian blur (sigma=1), zero padding, LDS tile 20x20
// ---------------------------------------------------------------------------
__global__ __launch_bounds__(256) void blur5(
    const float* __restrict__ in, float* __restrict__ out)
{
    __shared__ float tile[20][20];
    const int im = blockIdx.z;
    const int tx = threadIdx.x, ty = threadIdx.y;
    const int x0 = blockIdx.x * 16, y0 = blockIdx.y * 16;
    const float* src = in + (size_t)im * IMG_PIX;

    for (int i = ty * 16 + tx; i < 400; i += 256) {
        const int ly = i / 20, lx = i % 20;
        const int gx = x0 + lx - 2, gy = y0 + ly - 2;
        float v = 0.0f;
        if (gx >= 0 && gx < NIMG && gy >= 0 && gy < NIMG)
            v = src[gy * NIMG + gx];
        tile[ly][lx] = v;
    }
    __syncthreads();

    // normalized 1D Gaussian (sigma=1, size=5): g/sum(g)
    const float k1[5] = { 0.05448868455f, 0.24420134200f, 0.40261994690f,
                          0.24420134200f, 0.05448868455f };
    float acc = 0.0f;
    #pragma unroll
    for (int i = 0; i < 5; ++i) {
        float rowAcc = 0.0f;
        #pragma unroll
        for (int j = 0; j < 5; ++j)
            rowAcc += k1[j] * tile[ty + i][tx + j];
        acc += k1[i] * rowAcc;
    }
    out[(size_t)im * IMG_PIX + (y0 + ty) * NIMG + (x0 + tx)] = acc;
}

// ---------------------------------------------------------------------------
// 256-point radix-2 DIT FFT in LDS (128 threads; data pre-permuted bit-rev).
// sign = -1 forward, +1 inverse (unnormalized).
// ---------------------------------------------------------------------------
static __device__ __forceinline__ void fft256(float2* s, int t, float sign)
{
    #pragma unroll
    for (int st = 0; st < 8; ++st) {
        __syncthreads();
        const int half = 1 << st;
        const int grp  = t >> st;
        const int pos  = t & (half - 1);
        const int i0   = (grp << (st + 1)) + pos;
        const int i1   = i0 + half;
        const float ang = sign * 3.14159265358979323846f * (float)pos / (float)half;
        float sn, cs;
        __sincosf(ang, &sn, &cs);
        const float2 a = s[i0];
        const float2 b = s[i1];
        const float2 tb = make_float2(b.x * cs - b.y * sn, b.x * sn + b.y * cs);
        s[i0] = make_float2(a.x + tb.x, a.y + tb.y);
        s[i1] = make_float2(a.x - tb.x, a.y - tb.y);
    }
    __syncthreads();
}

static __device__ __forceinline__ int brev8(int x) {
    return (int)(__brev((unsigned)x) >> 24);
}

// Pass A: forward FFT along rows (real input -> complex buffer)
__global__ __launch_bounds__(128) void fft_row_fwd(
    const float* __restrict__ in, float2* __restrict__ cbuf)
{
    __shared__ float2 s[256];
    const int line = blockIdx.x;            // img*256 + row
    const int t = threadIdx.x;
    const float* src = in + (size_t)line * NIMG;
    for (int e = t; e < NIMG; e += 128)
        s[brev8(e)] = make_float2(src[e], 0.0f);
    fft256(s, t, -1.0f);
    float2* dst = cbuf + (size_t)line * NIMG;
    for (int e = t; e < NIMG; e += 128) dst[e] = s[e];
}

// Pass B: forward FFT along columns, multiply by real CTF, inverse column FFT
__global__ __launch_bounds__(128) void fft_col_ctf(
    float2* __restrict__ cbuf, const float* __restrict__ ctf)
{
    __shared__ float2 s[256];
    const int col = blockIdx.x;
    const int im  = blockIdx.y;
    const int t   = threadIdx.x;
    float2* base = cbuf + (size_t)im * IMG_PIX + col;

    for (int e = t; e < NIMG; e += 128)
        s[brev8(e)] = base[(size_t)e * NIMG];
    fft256(s, t, -1.0f);

    // multiply by ctf[im][ky][col] (each thread owns e = t, t+128)
    for (int e = t; e < NIMG; e += 128) {
        const float f = ctf[(size_t)im * IMG_PIX + e * NIMG + col];
        s[e].x *= f; s[e].y *= f;
    }
    // re-permute (bit-reverse) for the inverse transform
    const float2 v0 = s[t];
    const float2 v1 = s[t + 128];
    __syncthreads();
    s[brev8(t)]       = v0;
    s[brev8(t + 128)] = v1;
    fft256(s, t, +1.0f);

    for (int e = t; e < NIMG; e += 128)
        base[(size_t)e * NIMG] = s[e];
}

// Pass C: inverse FFT along rows, take real part, scale 1/(N*N)
__global__ __launch_bounds__(128) void fft_row_inv(
    const float2* __restrict__ cbuf, float* __restrict__ out)
{
    __shared__ float2 s[256];
    const int line = blockIdx.x;
    const int t = threadIdx.x;
    const float2* src = cbuf + (size_t)line * NIMG;
    for (int e = t; e < NIMG; e += 128)
        s[brev8(e)] = src[e];
    fft256(s, t, +1.0f);
    float* dst = out + (size_t)line * NIMG;
    const float scale = 1.0f / (float)(NIMG * NIMG);
    for (int e = t; e < NIMG; e += 128) dst[e] = s[e].x * scale;
}

// ---------------------------------------------------------------------------
// Host launcher
// ---------------------------------------------------------------------------
extern "C" void kernel_launch(void* const* d_in, const int* in_sizes, int n_in,
                              void* d_out, int out_size, void* d_ws, size_t ws_size,
                              hipStream_t stream)
{
    (void)in_sizes; (void)n_in; (void)out_size; (void)ws_size;

    const float* zx     = (const float*)d_in[0];
    const float* zy     = (const float*)d_in[1];
    const float* zz     = (const float*)d_in[2];
    const float* Zb     = (const float*)d_in[3];
    const float* coords = (const float*)d_in[4];
    const float* wts    = (const float*)d_in[5];
    const float* Rm     = (const float*)d_in[6];
    const float* shifts = (const float*)d_in[7];
    const float* ctf    = (const float*)d_in[8];
    float* out = (float*)d_out;

    // workspace layout: [accum 8MB][blur 8MB][complex 16MB]
    float*  imgAcc = (float*)d_ws;
    float*  imgBlr = imgAcc + (size_t)NB * IMG_PIX;
    float2* cbuf   = (float2*)((char*)d_ws + 2ull * NB * IMG_PIX * sizeof(float));

    const int totalPix = NB * IMG_PIX;                       // 2,097,152
    zero_buf<<<(totalPix + 255) / 256, 256, 0, stream>>>(imgAcc, totalPix);

    const int blocks = (TILES + 7) / 8;                      // 8 waves/block
    splat_wmma<<<blocks, 256, 0, stream>>>(zx, zy, zz, Zb, coords, wts,
                                           Rm, shifts, imgAcc);

    blur5<<<dim3(NIMG / 16, NIMG / 16, NB), dim3(16, 16), 0, stream>>>(imgAcc, imgBlr);

    fft_row_fwd<<<NB * NIMG, 128, 0, stream>>>(imgBlr, cbuf);
    fft_col_ctf<<<dim3(NIMG, NB), 128, 0, stream>>>(cbuf, ctf);
    fft_row_inv<<<NB * NIMG, 128, 0, stream>>>(cbuf, out);
}
